// RNNLayer_56753697849805
// MI455X (gfx1250) — compile-verified, matrix operand order checked by reading, not run on hardware
//
#include <hip/hip_runtime.h>

// ---------------------------------------------------------------------------
// Elman RNN for MI455X (gfx1250, wave32, WMMA bf16 16x16x32, fp32 accumulate)
//   T=256, B=128, H=2048
//   xproj = x @ w_in + bias   (parallel GEMM, bf16 output in ws)
//   h_t   = tanh(xproj_t + h_{t-1} @ w_rec)   (256 sequential step kernels)
// ---------------------------------------------------------------------------

#define RNN_T 256
#define RNN_B 128
#define RNN_H 2048
#define KT    (RNN_H / 32)   // 64 K-tiles of 32

typedef __attribute__((ext_vector_type(16))) __bf16 v16bf;
typedef __attribute__((ext_vector_type(8)))  __bf16 v8bf;
typedef __attribute__((ext_vector_type(8)))  float  v8f;

union AFrag { v16bf v; v8bf h[2]; };

// ---------------------------------------------------------------------------
// Weight swizzle: fp32 row-major (K x N) -> bf16 in WMMA B-operand lane order.
// Tile (kt, nt) of 32(K) x 16(N). Within a tile: lane l, slot j (0..15):
//   N = nt*16 + (l & 15)
//   K = kt*32 + j + 8*(j>>3) + 8*(l>>4)
// stored contiguously as dst[tile*512 + l*16 + j]  (one v16bf per lane).
// One thread per bf16 *pair* (j even) -> packed 32-bit store.
// ---------------------------------------------------------------------------
__global__ __launch_bounds__(256) void swizzle_w_kernel(
    const float* __restrict__ w_in, const float* __restrict__ w_rec,
    __bf16* __restrict__ dst_in, __bf16* __restrict__ dst_rec)
{
    const float* __restrict__ src = blockIdx.y ? w_rec : w_in;
    __bf16*      __restrict__ dst = blockIdx.y ? dst_rec : dst_in;

    unsigned p    = blockIdx.x * 256u + threadIdx.x;  // pair index, H*H/2 total
    int j         = (p & 7) * 2;                      // even slot
    int l         = (p >> 3) & 31;                    // lane
    unsigned tile = p >> 8;                           // kt + nt*KT packed
    int kt        = tile & (KT - 1);
    int nt        = tile >> 6;                        // H/16 = 128 values
    int n         = nt * 16 + (l & 15);
    int K0        = kt * 32 + j + 8 * (j >> 3) + 8 * (l >> 4);

    float f0 = src[(size_t)K0 * RNN_H + n];
    float f1 = src[(size_t)(K0 + 1) * RNN_H + n];
    union { unsigned u; __bf16 b[2]; } pk;
    pk.b[0] = (__bf16)f0;
    pk.b[1] = (__bf16)f1;
    ((unsigned*)dst)[(size_t)tile * 256 + l * 8 + (j >> 1)] = pk.u;
}

// ---------------------------------------------------------------------------
// Gather this lane's A fragment (16x32 bf16 tile staged row-major in LDS).
// lane<16:  K in [0,8) U [16,24);  lane>=16: K in [8,16) U [24,32); M = lane&15.
// Two 16-byte ds loads per lane.
// ---------------------------------------------------------------------------
__device__ inline v16bf gather_a(const __bf16* atile, int lane)
{
    const int lm = lane & 15, hi = lane >> 4;
    const __bf16* ar = atile + lm * 32 + hi * 8;
    AFrag af;
    af.h[0] = *(const v8bf*)(ar);
    af.h[1] = *(const v8bf*)(ar + 16);
    return af.v;
}

__device__ inline const v16bf* bptr(const __bf16* wsw, int nt, int kt, int lane)
{
    return (const v16bf*)(wsw + (((size_t)nt * KT + kt) << 9) + lane * 16);
}

// ---------------------------------------------------------------------------
// xproj GEMM: (T*B x H) fp32 A  x  swizzled bf16 w_in  + bias -> bf16 xproj.
// Block: 256 thr = 8 waves, block tile 16(M) x 512(N); wave tile 16 x 64.
// Grid: (T*B/16) * (H/512) = 2048 * 4 = 8192 blocks.
// ---------------------------------------------------------------------------
__global__ __launch_bounds__(256) void xproj_kernel(
    const float* __restrict__ x, const __bf16* __restrict__ wsw,
    const float* __restrict__ bias, __bf16* __restrict__ xp)
{
    __shared__ __align__(32) __bf16 atile[16 * 32];

    const int lane = threadIdx.x & 31;
    const int wid  = threadIdx.x >> 5;     // 0..7
    const int mt   = blockIdx.x >> 2;      // 0..2047
    const int ng   = blockIdx.x & 3;       // 0..3
    const int m0   = mt * 16;
    const int n0   = ng * 512 + wid * 64;
    const int lm   = lane & 15, hi = lane >> 4;
    const int nt0  = n0 >> 4;

    v8f acc0, acc1, acc2, acc3;
    {
        float b0 = bias[n0 + lm];
        float b1 = bias[n0 + 16 + lm];
        float b2 = bias[n0 + 32 + lm];
        float b3 = bias[n0 + 48 + lm];
        acc0 = (v8f){b0, b0, b0, b0, b0, b0, b0, b0};
        acc1 = (v8f){b1, b1, b1, b1, b1, b1, b1, b1};
        acc2 = (v8f){b2, b2, b2, b2, b2, b2, b2, b2};
        acc3 = (v8f){b3, b3, b3, b3, b3, b3, b3, b3};
    }

    // cooperative A staging: 256 threads * 2 elems = 16x32 tile
    const int se   = threadIdx.x * 2;
    const int srow = se >> 5;
    const int scol = se & 31;
    const float* __restrict__ srcrow = x + (size_t)(m0 + srow) * RNN_H + scol;

#pragma unroll 1
    for (int kt = 0; kt < KT; ++kt) {
        __syncthreads();                       // WAR on atile
        {
            const float* sp = srcrow + kt * 32;
            if (kt + 1 < KT) __builtin_prefetch(sp + 32, 0, 1);
            float2 f = *(const float2*)sp;
            union { unsigned u; __bf16 b[2]; } pk;
            pk.b[0] = (__bf16)f.x;
            pk.b[1] = (__bf16)f.y;
            ((unsigned*)atile)[(srow * 32 + scol) >> 1] = pk.u;
        }
        __syncthreads();                       // RAW on atile

        const v16bf a  = gather_a(atile, lane);
        // issue all 4 B loads first (independent regs -> one clause),
        // then 4 back-to-back WMMAs (no D->A/B hazard between them)
        const v16bf b0 = *bptr(wsw, nt0 + 0, kt, lane);
        const v16bf b1 = *bptr(wsw, nt0 + 1, kt, lane);
        const v16bf b2 = *bptr(wsw, nt0 + 2, kt, lane);
        const v16bf b3 = *bptr(wsw, nt0 + 3, kt, lane);
        acc0 = __builtin_amdgcn_wmma_f32_16x16x32_bf16(false, a, false, b0,
                                                       (short)0, acc0, false, false);
        acc1 = __builtin_amdgcn_wmma_f32_16x16x32_bf16(false, a, false, b1,
                                                       (short)0, acc1, false, false);
        acc2 = __builtin_amdgcn_wmma_f32_16x16x32_bf16(false, a, false, b2,
                                                       (short)0, acc2, false, false);
        acc3 = __builtin_amdgcn_wmma_f32_16x16x32_bf16(false, a, false, b3,
                                                       (short)0, acc3, false, false);
    }

    // epilogue: D layout -> VGPR r: M = r (lanes<16) / r+8 (lanes>=16), N = lm
    const v8f accs[4] = {acc0, acc1, acc2, acc3};
#pragma unroll
    for (int nn = 0; nn < 4; ++nn)
#pragma unroll
        for (int r = 0; r < 8; ++r) {
            int row = m0 + r + hi * 8;
            int col = n0 + nn * 16 + lm;
            xp[(size_t)row * RNN_H + col] = (__bf16)accs[nn][r];
        }
}

// ---------------------------------------------------------------------------
// One recurrent step: hout = tanh(xp_t + hin @ w_rec)
// Block: 128 thr = 4 waves, block tile 16 x 256; grid 8 * 8 = 64 blocks.
// w_rec (bf16, 8 MB) is L2-resident across all 256 steps.
// ---------------------------------------------------------------------------
__global__ __launch_bounds__(128) void step_kernel(
    const float* __restrict__ hin, const __bf16* __restrict__ wsw,
    const __bf16* __restrict__ xp_t, float* __restrict__ hout)
{
    __shared__ __align__(32) __bf16 atile[16 * 32];

    const int lane = threadIdx.x & 31;
    const int wid  = threadIdx.x >> 5;     // 0..3
    const int mt   = blockIdx.x >> 3;      // 0..7  (B/16)
    const int ng   = blockIdx.x & 7;       // 0..7  (H/256)
    const int m0   = mt * 16;
    const int n0   = ng * 256 + wid * 64;
    const int lm   = lane & 15, hi = lane >> 4;
    const int nt0  = n0 >> 4;

    v8f acc0 = (v8f){0, 0, 0, 0, 0, 0, 0, 0};
    v8f acc1 = acc0, acc2 = acc0, acc3 = acc0;

    // cooperative A staging: 128 threads * 4 elems = 16x32 tile
    const int se   = threadIdx.x * 4;
    const int srow = se >> 5;
    const int scol = se & 31;
    const float* __restrict__ srcrow = hin + (size_t)(m0 + srow) * RNN_H + scol;

#pragma unroll 1
    for (int kt = 0; kt < KT; ++kt) {
        __syncthreads();
        {
            const float4 f = *(const float4*)(srcrow + kt * 32);
            union { unsigned u; __bf16 b[2]; } p0, p1;
            p0.b[0] = (__bf16)f.x; p0.b[1] = (__bf16)f.y;
            p1.b[0] = (__bf16)f.z; p1.b[1] = (__bf16)f.w;
            unsigned* d = (unsigned*)atile + ((srow * 32 + scol) >> 1);
            d[0] = p0.u;
            d[1] = p1.u;
        }
        __syncthreads();

        const v16bf a  = gather_a(atile, lane);
        const v16bf b0 = *bptr(wsw, nt0 + 0, kt, lane);
        const v16bf b1 = *bptr(wsw, nt0 + 1, kt, lane);
        const v16bf b2 = *bptr(wsw, nt0 + 2, kt, lane);
        const v16bf b3 = *bptr(wsw, nt0 + 3, kt, lane);
        acc0 = __builtin_amdgcn_wmma_f32_16x16x32_bf16(false, a, false, b0,
                                                       (short)0, acc0, false, false);
        acc1 = __builtin_amdgcn_wmma_f32_16x16x32_bf16(false, a, false, b1,
                                                       (short)0, acc1, false, false);
        acc2 = __builtin_amdgcn_wmma_f32_16x16x32_bf16(false, a, false, b2,
                                                       (short)0, acc2, false, false);
        acc3 = __builtin_amdgcn_wmma_f32_16x16x32_bf16(false, a, false, b3,
                                                       (short)0, acc3, false, false);
    }

    const v8f accs[4] = {acc0, acc1, acc2, acc3};
#pragma unroll
    for (int nn = 0; nn < 4; ++nn)
#pragma unroll
        for (int r = 0; r < 8; ++r) {
            int row = m0 + r + hi * 8;
            int col = n0 + nn * 16 + lm;
            float v = accs[nn][r] + (float)xp_t[(size_t)row * RNN_H + col];
            hout[(size_t)row * RNN_H + col] = tanhf(v);
        }
}

// ---------------------------------------------------------------------------
// Host-side orchestration (all on `stream`, graph-capture safe).
// Workspace layout (bytes):
//   [0,   8M)        w_in  swizzled bf16
//   [8M,  16M)       w_rec swizzled bf16
//   [16M, 16M+134M)  xproj bf16 (T*B x H)
//   then two fp32 h ping-pong buffers (1 MB each)
// ---------------------------------------------------------------------------
extern "C" void kernel_launch(void* const* d_in, const int* in_sizes, int n_in,
                              void* d_out, int out_size, void* d_ws, size_t ws_size,
                              hipStream_t stream)
{
    const float* x     = (const float*)d_in[0];
    const float* w_in  = (const float*)d_in[1];
    const float* w_rec = (const float*)d_in[2];
    const float* bias  = (const float*)d_in[3];
    float*       out   = (float*)d_out;

    char* ws = (char*)d_ws;
    const size_t WSZ   = (size_t)RNN_H * RNN_H * 2;          // 8 MB
    const size_t XPOFF = 2 * WSZ;                            // 16 MB
    const size_t XPSZ  = (size_t)RNN_T * RNN_B * RNN_H * 2;  // 134 MB
    const size_t HSZ   = (size_t)RNN_B * RNN_H * 4;          // 1 MB

    __bf16* wsw_in  = (__bf16*)(ws);
    __bf16* wsw_rec = (__bf16*)(ws + WSZ);
    __bf16* xp      = (__bf16*)(ws + XPOFF);
    float*  hbuf0   = (float*)(ws + XPOFF + XPSZ);
    float*  hbuf1   = (float*)(ws + XPOFF + XPSZ + HSZ);

    // 1) swizzle both weight matrices to bf16 WMMA-B layout
    swizzle_w_kernel<<<dim3((RNN_H * RNN_H / 2) / 256, 2), 256, 0, stream>>>(
        w_in, w_rec, wsw_in, wsw_rec);

    // 2) h0 = 0
    hipMemsetAsync(hbuf0, 0, HSZ, stream);

    // 3) xproj = x @ w_in + bias (bf16)
    xproj_kernel<<<(RNN_T * RNN_B / 16) * (RNN_H / 512), 256, 0, stream>>>(
        x, wsw_in, bias, xp);

    // 4) 256 sequential recurrent steps, ping-ponging h buffers
    for (int t = 0; t < RNN_T; ++t) {
        const float* hin  = (t & 1) ? hbuf1 : hbuf0;
        float*       hout = (t == RNN_T - 1) ? out : ((t & 1) ? hbuf0 : hbuf1);
        const __bf16* xpt = xp + (size_t)t * RNN_B * RNN_H;
        step_kernel<<<(RNN_B / 16) * (RNN_H / 256), 128, 0, stream>>>(
            hin, wsw_rec, xpt, hout);
    }
}
